// BiaffineModule_45311904973554
// MI455X (gfx1250) — matrix operand order repaired and checked by reading, not compiled, not run on hardware
//
#include <hip/hip_runtime.h>
#include <hip/hip_bf16.h>
#include <stddef.h>

// ---------------- constants (match reference shapes) ----------------
#define BB 4
#define SS 256
#define HH 768
#define FF 32
#define TT 256
#define WD 64
#define CC 16
#define KX (HH + FF)          // 800

typedef _Float16 half_t;
typedef __attribute__((ext_vector_type(16))) _Float16 v16h;
typedef __attribute__((ext_vector_type(8)))  _Float16 v8h;
typedef __attribute__((ext_vector_type(8)))  float    v8f;

// ---------------- fragment loaders (CDNA5 wave32 WMMA layouts) ----------------
__device__ __forceinline__ v8h load8(const half_t* p) {
  return *(const v8h*)p;
}
__device__ __forceinline__ v16h cat16(v8h lo, v8h hi) {
  return __builtin_shufflevector(lo, hi, 0,1,2,3,4,5,6,7,8,9,10,11,12,13,14,15);
}

// A-matrix 16x32 f16 fragment, row-major source (ld halves).
// Lanes 0-15: M=lane, halves 0-7 -> K=k0..k0+7, halves 8-15 -> K=k0+16..k0+23.
// Lanes 16-31: same M, K offset by +8.
__device__ __forceinline__ v16h load_fragA(const half_t* base, int ld,
                                           int row0, int k0, int lane) {
  int m  = lane & 15;
  int kb = k0 + ((lane & 16) ? 8 : 0);
  const half_t* p = base + (size_t)(row0 + m) * ld + kb;
  return cat16(load8(p), load8(p + 16));
}

// B-matrix 32x16 f16 fragment where B[k][n] = src[n][k] (src row-major, ld halves).
// Lanes 0-15: N=lane, halves 0-7 -> K=k0..k0+7, halves 8-15 -> K=k0+16..k0+23.  Lanes 16-31: K offset by +8.
__device__ __forceinline__ v16h load_fragB(const half_t* base, int ld,
                                           int col0, int k0, int lane) {
  int n  = col0 + (lane & 15);
  int kb = k0 + ((lane & 16) ? 8 : 0);
  const half_t* p = base + (size_t)n * ld + kb;
  return cat16(load8(p), load8(p + 16));
}

// 64x32 (4x2 sub-tile) register-blocked wave GEMM core.
// A: row-major M x K (lda), Bt: row-major N x K (ldb) i.e. B^T -> B[k][n]=Bt[n][k].
// 8 WMMAs per 6 fragment loads per k-step (0.75 frags/WMMA vs 2.0 unblocked).
__device__ __forceinline__ void wave_mma_64x32(const half_t* __restrict__ A, int lda,
                                               const half_t* __restrict__ Bt, int ldb,
                                               int mBase, int nBase, int K,
                                               int lane, v8f acc[4][2]) {
  const v8f vzero = {};
  #pragma unroll
  for (int i = 0; i < 4; ++i)
    #pragma unroll
    for (int j = 0; j < 2; ++j)
      acc[i][j] = vzero;

  for (int k = 0; k < K; k += 32) {
    v16h a[4], b[2];
    #pragma unroll
    for (int i = 0; i < 4; ++i)
      a[i] = load_fragA(A, lda, mBase + 16 * i, k, lane);
    #pragma unroll
    for (int j = 0; j < 2; ++j)
      b[j] = load_fragB(Bt, ldb, nBase + 16 * j, k, lane);
    #pragma unroll
    for (int i = 0; i < 4; ++i)
      #pragma unroll
      for (int j = 0; j < 2; ++j)
        acc[i][j] = __builtin_amdgcn_wmma_f32_16x16x32_f16(
            false, a[i], false, b[j], (short)0, acc[i][j], false, false);
  }
}

// ---------------- prep: fp32 -> f16 conversions / transposes ----------------
__global__ void prep_x16(const float* __restrict__ fh, const float* __restrict__ fv,
                         half_t* __restrict__ x16) {
  int idx = blockIdx.x * blockDim.x + threadIdx.x;     // over B*S*KX
  if (idx >= BB * SS * KX) return;
  int row = idx / KX, col = idx - row * KX;
  float v = (col < HH) ? fh[(size_t)row * HH + col]
                       : fv[(size_t)row * FF + (col - HH)];
  x16[idx] = (half_t)v;
}

// wT[(which*T + n)*KX + k] = W_which[k*T + n]   (f16, transposed)
__global__ void prep_wT(const float* __restrict__ startW, const float* __restrict__ endW,
                        half_t* __restrict__ wT) {
  int idx = blockIdx.x * blockDim.x + threadIdx.x;     // over 2*T*KX
  if (idx >= 2 * TT * KX) return;
  int which = idx / (TT * KX);
  int r = idx - which * (TT * KX);
  int n = r / KX, k = r - n * KX;
  const float* W = which ? endW : startW;
  wT[idx] = (half_t)W[(size_t)k * TT + n];
}

// uT[(c*T + n)*T + t] = U[t, c, n]   (f16, per-channel transposed)
__global__ void prep_uT(const float* __restrict__ U, half_t* __restrict__ uT) {
  int idx = blockIdx.x * blockDim.x + threadIdx.x;     // over C*T*T
  if (idx >= CC * TT * TT) return;
  int c = idx / (TT * TT);
  int r = idx - c * (TT * TT);
  int n = r / TT, t = r - n * TT;
  uT[idx] = (half_t)U[((size_t)t * CC + c) * TT + n];
}

// ---------------- GEMM 1: s,e = x @ W + b  (M=B*S=1024, N=256, K=800) -------
__global__ void gemm_se_kernel(const half_t* __restrict__ x16, const half_t* __restrict__ wT,
                               const float* __restrict__ start_b, const float* __restrict__ end_b,
                               float* __restrict__ s32, float* __restrict__ e32,
                               half_t* __restrict__ s16, half_t* __restrict__ e16) {
  int lane  = threadIdx.x & 31;
  int wave  = threadIdx.x >> 5;
  int mBase = (blockIdx.x * 4 + wave) * 64;   // 1024 rows / (4 blk * 4 waves) = 64
  int nBase = blockIdx.y * 32;                // 256 cols / 8 = 32
  int which = blockIdx.z;                     // 0: start, 1: end
  const half_t* W = wT + (size_t)which * TT * KX;

  v8f acc[4][2];
  wave_mma_64x32(x16, KX, W, KX, mBase, nBase, KX, lane, acc);

  const float* bias = which ? end_b : start_b;
  float*  o32 = which ? e32 : s32;
  half_t* o16 = which ? e16 : s16;
  #pragma unroll
  for (int j = 0; j < 2; ++j) {
    int col = nBase + 16 * j + (lane & 15);
    float bv = bias[col];
    #pragma unroll
    for (int i = 0; i < 4; ++i) {
      int rb = mBase + 16 * i + ((lane & 16) ? 8 : 0);
      #pragma unroll
      for (int r = 0; r < 8; ++r) {
        float v = acc[i][j][r] + bv;
        o32[(size_t)(rb + r) * TT + col] = v;
        o16[(size_t)(rb + r) * TT + col] = (half_t)v;
      }
    }
  }
}

// ---------------- small FFN terms (rank-16, VALU) ----------------
__global__ void ffn_se_kernel(const float* __restrict__ s32, const float* __restrict__ e32,
                              const float* __restrict__ linW,
                              float* __restrict__ ffn_s, float* __restrict__ ffn_e) {
  int idx = blockIdx.x * blockDim.x + threadIdx.x;     // over B*S*C
  if (idx >= BB * SS * CC) return;
  int row = idx >> 4, c = idx & 15;
  float as = 0.f, ae = 0.f;
  for (int t = 0; t < TT; ++t) {
    as += s32[(size_t)row * TT + t] * linW[(size_t)t * CC + c];
    ae += e32[(size_t)row * TT + t] * linW[(size_t)(TT + t) * CC + c];
  }
  ffn_s[idx] = as;
  ffn_e[idx] = ae;
}

__global__ void ffn_w_kernel(const float* __restrict__ width, const float* __restrict__ linW,
                             float* __restrict__ ffn_w) {
  int idx = blockIdx.x * blockDim.x + threadIdx.x;     // over S*S*C
  if (idx >= SS * SS * CC) return;
  int pair = idx >> 4, c = idx & 15;
  float a = 0.f;
  #pragma unroll 8
  for (int w = 0; w < WD; ++w)
    a += width[(size_t)pair * WD + w] * linW[(size_t)(2 * TT + w) * CC + c];
  ffn_w[idx] = a;
}

// ---------------- GEMM 2: sU[b,c] = s_b @ U_c  (256x256x256 per pair) ------
__global__ void gemm_sU_kernel(const half_t* __restrict__ s16, const half_t* __restrict__ uT,
                               half_t* __restrict__ sU16) {
  int lane  = threadIdx.x & 31;
  int wave  = threadIdx.x >> 5;
  int mBase = wave * 64;        // 256 rows / 4 waves
  int nBase = blockIdx.y * 32;  // 256 cols / 8
  int bc = blockIdx.z;          // b*16 + c
  int b = bc >> 4, c = bc & 15;
  const half_t* A  = s16 + (size_t)b * TT * TT;
  const half_t* Bt = uT  + (size_t)c * TT * TT;

  v8f acc[4][2];
  wave_mma_64x32(A, TT, Bt, TT, mBase, nBase, TT, lane, acc);

  half_t* o = sU16 + (size_t)bc * TT * TT;
  #pragma unroll
  for (int j = 0; j < 2; ++j) {
    int col = nBase + 16 * j + (lane & 15);
    #pragma unroll
    for (int i = 0; i < 4; ++i) {
      int rb = mBase + 16 * i + ((lane & 16) ? 8 : 0);
      #pragma unroll
      for (int r = 0; r < 8; ++r)
        o[(size_t)(rb + r) * TT + col] = (half_t)acc[i][j][r];
    }
  }
}

// ---------------- GEMM 3 + epilogue: out = sU @ e^T + ffn ------------------
__global__ void gemm_pre_kernel(const half_t* __restrict__ sU16, const half_t* __restrict__ e16,
                                const float* __restrict__ ffn_s, const float* __restrict__ ffn_e,
                                const float* __restrict__ ffn_w, const float* __restrict__ lin_b,
                                float* __restrict__ out) {
  int lane  = threadIdx.x & 31;
  int wave  = threadIdx.x >> 5;
  int mBase = wave * 64;        // s rows
  int nBase = blockIdx.y * 32;  // e cols
  int bc = blockIdx.z;          // b*16 + c
  int b = bc >> 4, c = bc & 15;
  const half_t* A = sU16 + (size_t)bc * TT * TT;
  const half_t* E = e16  + (size_t)b  * TT * TT;   // B[k][n] = E[n][k] -> free transpose

  v8f acc[4][2];
  wave_mma_64x32(A, TT, E, TT, mBase, nBase, TT, lane, acc);

  float bias = lin_b[c];
  #pragma unroll
  for (int j = 0; j < 2; ++j) {
    int col = nBase + 16 * j + (lane & 15);          // e index
    float fe = ffn_e[((size_t)b * SS + col) * CC + c] + bias;
    #pragma unroll
    for (int i = 0; i < 4; ++i) {
      int rb = mBase + 16 * i + ((lane & 16) ? 8 : 0);
      #pragma unroll
      for (int r = 0; r < 8; ++r) {
        int srow = rb + r;
        float v = acc[i][j][r]
                + ffn_s[((size_t)b * SS + srow) * CC + c]
                + fe
                + ffn_w[((size_t)srow * SS + col) * CC + c];
        out[(((size_t)b * SS + srow) * SS + col) * CC + c] = v;
      }
    }
  }
}

// ---------------- host side ----------------
static inline size_t align256(size_t x) { return (x + 255) & ~(size_t)255; }

extern "C" void kernel_launch(void* const* d_in, const int* in_sizes, int n_in,
                              void* d_out, int out_size, void* d_ws, size_t ws_size,
                              hipStream_t stream) {
  const float* final_hidden = (const float*)d_in[0];   // (B,S,H)
  const float* feature_vecs = (const float*)d_in[1];   // (B,S,F)
  const float* start_W      = (const float*)d_in[2];   // (H+F,T)
  const float* start_b      = (const float*)d_in[3];   // (T)
  const float* end_W        = (const float*)d_in[4];   // (H+F,T)
  const float* end_b        = (const float*)d_in[5];   // (T)
  const float* U            = (const float*)d_in[6];   // (T,C,T)
  const float* width_hidden = (const float*)d_in[7];   // (S,S,WD)
  const float* lin_W        = (const float*)d_in[8];   // (2T+WD,C)
  const float* lin_b        = (const float*)d_in[9];   // (C)
  float* out = (float*)d_out;                          // (B,S,S,C)

  // workspace carve-up
  char* ws = (char*)d_ws;
  size_t off = 0;
  half_t* x16  = (half_t*)(ws + off); off = align256(off + sizeof(half_t) * BB * SS * KX);
  half_t* wT   = (half_t*)(ws + off); off = align256(off + sizeof(half_t) * 2  * TT * KX);
  half_t* uT   = (half_t*)(ws + off); off = align256(off + sizeof(half_t) * CC * TT * TT);
  float*  s32  = (float*) (ws + off); off = align256(off + sizeof(float)  * BB * SS * TT);
  float*  e32  = (float*) (ws + off); off = align256(off + sizeof(float)  * BB * SS * TT);
  half_t* s16  = (half_t*)(ws + off); off = align256(off + sizeof(half_t) * BB * SS * TT);
  half_t* e16  = (half_t*)(ws + off); off = align256(off + sizeof(half_t) * BB * SS * TT);
  half_t* sU16 = (half_t*)(ws + off); off = align256(off + sizeof(half_t) * (size_t)BB * CC * TT * TT);
  float*  ffnS = (float*) (ws + off); off = align256(off + sizeof(float)  * BB * SS * CC);
  float*  ffnE = (float*) (ws + off); off = align256(off + sizeof(float)  * BB * SS * CC);
  float*  ffnW = (float*) (ws + off); off = align256(off + sizeof(float)  * (size_t)SS * SS * CC);
  (void)ws_size; (void)in_sizes; (void)n_in; (void)out_size;

  // prep conversions (f32 -> f16, weights transposed for contiguous B fragments)
  {
    int n = BB * SS * KX;
    prep_x16<<<(n + 255) / 256, 256, 0, stream>>>(final_hidden, feature_vecs, x16);
  }
  {
    int n = 2 * TT * KX;
    prep_wT<<<(n + 255) / 256, 256, 0, stream>>>(start_W, end_W, wT);
  }
  {
    int n = CC * TT * TT;
    prep_uT<<<(n + 255) / 256, 256, 0, stream>>>(U, uT);
  }

  // s,e = x @ W + b   (WMMA f16->f32, 64x32 wave tiles)
  gemm_se_kernel<<<dim3(4, 8, 2), 128, 0, stream>>>(x16, wT, start_b, end_b,
                                                    s32, e32, s16, e16);

  // small FFN terms
  ffn_se_kernel<<<(BB * SS * CC + 255) / 256, 256, 0, stream>>>(s32, e32, lin_W, ffnS, ffnE);
  ffn_w_kernel<<<(SS * SS * CC + 255) / 256, 256, 0, stream>>>(width_hidden, lin_W, ffnW);

  // sU[b,c] = s_b @ U_c     (64 pairs of 256^3 WMMA GEMMs, f16 out -> L2 resident)
  gemm_sU_kernel<<<dim3(1, 8, 64), 128, 0, stream>>>(s16, uT, sU16);

  // out = sU @ e^T + ffn_s + ffn_e + ffn_w + lin_b   (fused epilogue)
  gemm_pre_kernel<<<dim3(1, 8, 64), 128, 0, stream>>>(sU16, e16, ffnS, ffnE, ffnW, lin_b, out);
}
